// Model_FN_26783416058402
// MI455X (gfx1250) — compile-verified, matrix operand order checked by reading, not run on hardware
//
#include <hip/hip_runtime.h>
#include <math.h>

// ---------------- constants from the reference ----------------
#define A_C    0.7f
#define B_C    0.8f
#define EPI_   0.2f
#define TAU_   0.1f
#define I_C    1.0f
#define UMAX_  10.0f
#define DSM_   0.1f
#define EPSR_  0.001f

#define NB    4096      // batch N
#define DIMD  1024      // dim
#define DIN   2048      // 2*dim
#define HID   512
#define LDO   2049      // output row stride (2048 dx + 1 V)

// ---------------- CDNA5 WMMA types ----------------
typedef __attribute__((ext_vector_type(16))) __bf16       v16bf;
typedef __attribute__((ext_vector_type(8)))  float        v8f;
typedef __attribute__((ext_vector_type(4)))  unsigned int u32x4;
typedef __attribute__((ext_vector_type(4)))  int          i32x4;

union FragBF { v16bf v; u32x4 q[2]; };

// ---------------- CDNA5 async global->LDS path (with fallback) ----------------
#ifndef __has_builtin
#define __has_builtin(x) 0
#endif
#if __has_builtin(__builtin_amdgcn_global_load_async_to_lds_b128)
#define ASYNC_LDS 1
// builtin signature (from toolchain diagnostic): (int4 AS1*, int4 AS3*, imm, imm)
#define GPTR(p) ((__attribute__((address_space(1))) i32x4*)(p))
#define LPTR(p) ((__attribute__((address_space(3))) i32x4*)(p))
#else
#define ASYNC_LDS 0
#endif

#if __has_builtin(__builtin_amdgcn_s_wait_asynccnt)
#define WAIT_ASYNC(n) __builtin_amdgcn_s_wait_asynccnt(n)
#else
#define WAIT_ASYNC(n) asm volatile("s_wait_asynccnt %0" ::"i"(n) : "memory")
#endif

// ---------------- helpers ----------------
__device__ __forceinline__ unsigned short f2bf(float f) {
  unsigned int u = __float_as_uint(f);
  unsigned int r = u + 0x7FFFu + ((u >> 16) & 1u);   // RNE
  return (unsigned short)(r >> 16);
}
__device__ __forceinline__ float bf2f(unsigned short h) {
  return __uint_as_float(((unsigned int)h) << 16);
}
__device__ __forceinline__ float smooth_relu_f(float x) {
  float r  = fmaxf(x, 0.f);
  float sq = (2.f * DSM_ * r * r * r - r * r * r * r) * (1.f / (2.f * DSM_ * DSM_ * DSM_));
  return (r < DSM_) ? sq : (x - 0.5f * DSM_);
}
__device__ __forceinline__ float softplus_f(float x) { return log1pf(expf(x)); }

// ---------------- epilogue modes ----------------
#define M_STORE_BF16   0   // outb = bf(acc)                     (Lm)
#define M_ADD_OUT      1   // outf += acc                        (du += Hu@Lm.T)
#define M_TANH_BF16    2   // outb = bf(tanh(acc+bias))          (h1,h2)
#define M_CTRL         3   // clamp((acc+bias)*st) added to outf (control)
#define M_SR_BF16      4   // outb = bf(smooth_relu(acc+bias))   (Z1)
#define M_STORE_F32B   5   // outf = acc+bias                    (T2)
#define M_ADDC_SR_F32  6   // outf = smooth_relu(acc+Cin)        (Z2)

// ============================================================================
// Tiled bf16 WMMA GEMM:  out[m][n] = epilogue( sum_k A[m][k] * B[n][k] )
// A: [M x K] bf16 row-major (lda), B: [N x K] bf16 row-major (ldb).
// Block = BM x BN, 256 threads (8 waves, 2x4), BK = 32.
// Double-buffered LDS, async global->LDS (ASYNCcnt) pipeline:
//   issue tile k+1 -> s_wait_asynccnt CH -> barrier -> wmma tile k -> barrier
// All problem dims are multiples of the tile sizes -> no bounds checks.
// ============================================================================
template <int BM, int BN, int MODE>
__global__ __launch_bounds__(256) void gemm_bf16(
    const unsigned short* __restrict__ A, int lda,
    const unsigned short* __restrict__ B, int ldb,
    int K,
    const float* __restrict__ bias,
    const float* __restrict__ Cin, int ldcin,
    const unsigned short* __restrict__ staux, int ldaux,
    float* __restrict__ outf, unsigned short* __restrict__ outb,
    int ldc, int col_off)
{
  constexpr int MT  = BM / 2 / 16;      // m-tiles per wave (2 wave rows)
  constexpr int NT  = BN / 4 / 16;      // n-tiles per wave (4 wave cols)
  constexpr int CHA = (BM * 4) / 256;   // 16B chunks per thread for A tile
  constexpr int CHB = (BN * 4) / 256;   // 16B chunks per thread for B tile
  constexpr int CH  = CHA + CHB;        // async ops in flight per prefetched tile

  __shared__ unsigned short As[2][BM][32];
  __shared__ unsigned short Bs[2][BN][32];

  const int tid  = threadIdx.x;
  const int wave = tid >> 5;
  const int lane = tid & 31;
  const int wm   = wave >> 2;           // 0..1
  const int wn   = wave & 3;            // 0..3
  const int hlf  = lane >> 4;           // lane half (K/M split per ISA layout)
  const int l16  = lane & 15;

  const int block_m = blockIdx.y * BM;
  const int block_n = blockIdx.x * BN;

  v8f acc[MT][NT] = {};

  // issue one tile's global->LDS copies (async on CDNA5, sync fallback)
  auto issue = [&](int kt, int buf) {
    const int k0 = kt << 5;
#pragma unroll
    for (int i = 0; i < CHA; ++i) {
      int c = tid + i * 256;
      int row = c >> 2, cq = c & 3;
#if ASYNC_LDS
      __builtin_amdgcn_global_load_async_to_lds_b128(
          GPTR(&A[(block_m + row) * lda + k0 + cq * 8]),
          LPTR(&As[buf][row][cq * 8]), 0, 0);
#else
      *(u32x4*)&As[buf][row][cq * 8] =
          *(const u32x4*)&A[(block_m + row) * lda + k0 + cq * 8];
#endif
    }
#pragma unroll
    for (int i = 0; i < CHB; ++i) {
      int c = tid + i * 256;
      int row = c >> 2, cq = c & 3;
#if ASYNC_LDS
      __builtin_amdgcn_global_load_async_to_lds_b128(
          GPTR(&B[(block_n + row) * ldb + k0 + cq * 8]),
          LPTR(&Bs[buf][row][cq * 8]), 0, 0);
#else
      *(u32x4*)&Bs[buf][row][cq * 8] =
          *(const u32x4*)&B[(block_n + row) * ldb + k0 + cq * 8];
#endif
    }
  };

  const int kTiles = K >> 5;
  issue(0, 0);
  for (int kt = 0; kt < kTiles; ++kt) {
    const int cur = kt & 1;
    const bool pre = (kt + 1 < kTiles);
    if (pre) issue(kt + 1, cur ^ 1);    // prefetch next tile into other buffer
#if ASYNC_LDS
    if (pre) WAIT_ASYNC(CH); else WAIT_ASYNC(0);  // current tile's copies done
#endif
    __syncthreads();                    // all waves' fills of `cur` visible

    // ---- LDS -> fragments (ISA 16-bit A/B layout: per-lane K halves) ----
    FragBF af[MT], bf_[NT];
#pragma unroll
    for (int mt = 0; mt < MT; ++mt) {
      int row = wm * (MT * 16) + mt * 16 + l16;
      af[mt].q[0] = *(const u32x4*)&As[cur][row][hlf * 8];
      af[mt].q[1] = *(const u32x4*)&As[cur][row][16 + hlf * 8];
    }
#pragma unroll
    for (int nt = 0; nt < NT; ++nt) {
      int row = wn * (NT * 16) + nt * 16 + l16;
      bf_[nt].q[0] = *(const u32x4*)&Bs[cur][row][hlf * 8];
      bf_[nt].q[1] = *(const u32x4*)&Bs[cur][row][16 + hlf * 8];
    }
#pragma unroll
    for (int mt = 0; mt < MT; ++mt)
#pragma unroll
      for (int nt = 0; nt < NT; ++nt)
        acc[mt][nt] = __builtin_amdgcn_wmma_f32_16x16x32_bf16(
            false, af[mt].v, false, bf_[nt].v, (short)0, acc[mt][nt],
            false, false);
    __syncthreads();                    // done reading `cur` before it is refilled
  }

  // ---- epilogue (C/D layout: VGPR r -> M = r + 8*half, N = lane%16) ----
#pragma unroll
  for (int mt = 0; mt < MT; ++mt) {
#pragma unroll
    for (int nt = 0; nt < NT; ++nt) {
      const int tm = block_m + wm * (MT * 16) + mt * 16;
      const int tn = block_n + wn * (NT * 16) + nt * 16;
#pragma unroll
      for (int r = 0; r < 8; ++r) {
        const int row = tm + r + 8 * hlf;
        const int col = tn + l16;
        float v = acc[mt][nt][r];
        if (MODE == M_STORE_BF16) {
          outb[row * ldc + col] = f2bf(v);
        } else if (MODE == M_ADD_OUT) {
          outf[row * ldc + col_off + col] += v;
        } else if (MODE == M_TANH_BF16) {
          outb[row * ldc + col] = f2bf(tanhf(v + bias[col]));
        } else if (MODE == M_CTRL) {
          float u  = v + bias[col];
          float x  = bf2f(staux[row * ldaux + col]);
          float ux = u * x;
          ux = (ux < UMAX_) ? ux : 0.f;
          ux = (-UMAX_ < ux) ? ux : 0.f;
          outf[row * ldc + col_off + col] += ux;
        } else if (MODE == M_SR_BF16) {
          outb[row * ldc + col] = f2bf(smooth_relu_f(v + bias[col]));
        } else if (MODE == M_STORE_F32B) {
          outf[row * ldc + col] = v + bias[col];
        } else if (MODE == M_ADDC_SR_F32) {
          outf[row * ldc + col] = smooth_relu_f(v + Cin[row * ldcin + col]);
        }
      }
    }
  }
}

// ============================================================================
// Elementwise prep: FHN terms, sigmoid(Hu), st, bf16 casts of state
// ============================================================================
__global__ void k_prep_state(const float* __restrict__ state,
                             float* __restrict__ out,
                             unsigned short* __restrict__ Hu,
                             unsigned short* __restrict__ stb,
                             unsigned short* __restrict__ stateb)
{
  int idx = blockIdx.x * blockDim.x + threadIdx.x;   // over NB*DIMD
  if (idx >= NB * DIMD) return;
  int m = idx >> 10, j = idx & (DIMD - 1);
  const float* row = state + m * DIN;
  float u = row[j], v = row[DIMD + j];
  float u0 = row[0], v0 = row[DIMD];
  Hu[m * DIMD + j] = f2bf(1.f / (1.f + expf(-u / TAU_)));
  float du = u - u * u * u * (1.f / 3.f) - v + I_C;
  float dv = EPI_ * (u + A_C - B_C * v);
  out[m * LDO + j]         = du;
  out[m * LDO + DIMD + j]  = dv;
  stb[m * DIN + j]         = f2bf(u - u0);
  stb[m * DIN + DIMD + j]  = f2bf(v - v0);
  stateb[m * DIN + j]        = f2bf(u);
  stateb[m * DIN + DIMD + j] = f2bf(v);
}

// ============================================================================
// Stage P -> PT (= V^T, 1024 x 1024, K padded) and PE (= (eig*V)^T)
// ============================================================================
__global__ void k_prep_P(const float* __restrict__ P,
                         const float* __restrict__ t,
                         unsigned short* __restrict__ PT,
                         unsigned short* __restrict__ PE)
{
  int idx = blockIdx.x * blockDim.x + threadIdx.x;   // over DIMD*DIMD
  if (idx >= DIMD * DIMD) return;
  int i = idx >> 10, r = idx & (DIMD - 1);
  float s = 2.f * sinf(t[0]);                        // K_=W_=1
  float val = (r < DIMD - 1) ? P[(1 + r) * DIMD + i] : 0.f;
  // eig index is i_eig = r+1; thresh = (dim+1)/2 - 1 = 511.5 -> i_eig <= 511
  float eig = ((r + 1) <= 511) ? (1.f + s) : (1.f - s);
  PT[i * DIMD + r] = f2bf(val);
  PE[i * DIMD + r] = f2bf(eig * val);
}

__global__ void k_cast_bf16(const float* __restrict__ s,
                            unsigned short* __restrict__ d, int n) {
  int i = blockIdx.x * blockDim.x + threadIdx.x;
  if (i < n) d[i] = f2bf(s[i]);
}
__global__ void k_softplus_bf16(const float* __restrict__ s,
                                unsigned short* __restrict__ d, int n) {
  int i = blockIdx.x * blockDim.x + threadIdx.x;
  if (i < n) d[i] = f2bf(softplus_f(s[i]));
}

// ============================================================================
// z0 = icnn(0): tiny 64-wide chain, one block
// ============================================================================
__global__ void k_z0(const float* __restrict__ icb0, const float* __restrict__ icb1,
                     const float* __restrict__ icb2, const float* __restrict__ icU0,
                     const float* __restrict__ icU1, float* __restrict__ z0out)
{
  __shared__ float z1[64], red[64];
  int j = threadIdx.x;
  z1[j] = smooth_relu_f(icb0[j]);
  __syncthreads();
  float a = icb1[j];
  for (int k = 0; k < 64; ++k) a += softplus_f(icU0[j * 64 + k]) * z1[k];
  red[j] = softplus_f(icU1[j]) * smooth_relu_f(a);
  __syncthreads();
  for (int off = 32; off; off >>= 1) {
    if (j < off) red[j] += red[j + off];
    __syncthreads();
  }
  if (j == 0) z0out[0] = icb2[0] + red[0];
}

// ============================================================================
// Final column: z[m] = state.icW2 + icb2 + sp(icU1).Z2 ; V = sr(z-z0)+eps*|x|^2
// One wave per row, wave32 shuffle reduction.
// ============================================================================
__global__ __launch_bounds__(256) void k_final(
    const float* __restrict__ state, const float* __restrict__ Z2f,
    const float* __restrict__ icW2, const float* __restrict__ icb2,
    const float* __restrict__ icU1, const float* __restrict__ z0p,
    float* __restrict__ out)
{
  int wave = threadIdx.x >> 5, lane = threadIdx.x & 31;
  int m = blockIdx.x * 8 + wave;
  if (m >= NB) return;
  const float* row = state + m * DIN;
  float dot = 0.f, ss = 0.f;
  for (int k = lane; k < DIN; k += 32) {
    float x = row[k];
    dot = fmaf(x, icW2[k], dot);
    ss  = fmaf(x, x, ss);
  }
  for (int j = lane; j < 64; j += 32)
    dot = fmaf(softplus_f(icU1[j]), Z2f[m * 64 + j], dot);
  for (int off = 16; off; off >>= 1) {
    dot += __shfl_xor(dot, off, 32);
    ss  += __shfl_xor(ss, off, 32);
  }
  if (lane == 0) {
    float z = dot + icb2[0];
    out[m * LDO + DIN] = smooth_relu_f(z - z0p[0]) + EPSR_ * ss;
  }
}

// ============================================================================
extern "C" void kernel_launch(void* const* d_in, const int* in_sizes, int n_in,
                              void* d_out, int out_size, void* d_ws, size_t ws_size,
                              hipStream_t stream) {
  const float* state = (const float*)d_in[0];
  const float* t     = (const float*)d_in[1];
  const float* P     = (const float*)d_in[2];
  const float* W1    = (const float*)d_in[3];
  const float* b1    = (const float*)d_in[4];
  const float* W2    = (const float*)d_in[5];
  const float* b2    = (const float*)d_in[6];
  const float* W3    = (const float*)d_in[7];
  const float* b3    = (const float*)d_in[8];
  const float* icW0  = (const float*)d_in[9];
  const float* icb0  = (const float*)d_in[10];
  const float* icW1  = (const float*)d_in[11];
  const float* icb1  = (const float*)d_in[12];
  const float* icW2  = (const float*)d_in[13];
  const float* icb2  = (const float*)d_in[14];
  const float* icU0  = (const float*)d_in[15];
  const float* icU1  = (const float*)d_in[16];
  float* out = (float*)d_out;

  char* ws = (char*)d_ws;
  size_t off = 0;
  auto alloc = [&](size_t bytes) -> void* {
    void* p = ws + off;
    off = (off + bytes + 255) & ~(size_t)255;
    return p;
  };
  typedef unsigned short us;
  us* stateB = (us*)alloc((size_t)NB * DIN * 2);
  us* stB    = (us*)alloc((size_t)NB * DIN * 2);
  us* HuB    = (us*)alloc((size_t)NB * DIMD * 2);
  us* LmB    = (us*)alloc((size_t)DIMD * DIMD * 2);
  us* PTB    = (us*)alloc((size_t)DIMD * DIMD * 2);
  us* PEB    = (us*)alloc((size_t)DIMD * DIMD * 2);
  us* W1B    = (us*)alloc((size_t)HID * DIN * 2);
  us* W2B    = (us*)alloc((size_t)HID * HID * 2);
  us* W3B    = (us*)alloc((size_t)DIN * HID * 2);
  us* h1B    = (us*)alloc((size_t)NB * HID * 2);
  us* h2B    = (us*)alloc((size_t)NB * HID * 2);
  us* icW0B  = (us*)alloc((size_t)64 * DIN * 2);
  us* icW1B  = (us*)alloc((size_t)64 * DIN * 2);
  us* spU0B  = (us*)alloc((size_t)64 * 64 * 2);
  us* Z1B    = (us*)alloc((size_t)NB * 64 * 2);
  float* T2f = (float*)alloc((size_t)NB * 64 * 4);
  float* Z2f = (float*)alloc((size_t)NB * 64 * 4);
  float* z0p = (float*)alloc(256);
  (void)ws_size; (void)in_sizes; (void)n_in; (void)out_size;

  const dim3 blk(256);

  // ---- prep ----
  k_prep_state<<<(NB * DIMD + 255) / 256, blk, 0, stream>>>(state, out, HuB, stB, stateB);
  k_prep_P<<<(DIMD * DIMD + 255) / 256, blk, 0, stream>>>(P, t, PTB, PEB);
  k_cast_bf16<<<(HID * DIN + 255) / 256, blk, 0, stream>>>(W1, W1B, HID * DIN);
  k_cast_bf16<<<(HID * HID + 255) / 256, blk, 0, stream>>>(W2, W2B, HID * HID);
  k_cast_bf16<<<(DIN * HID + 255) / 256, blk, 0, stream>>>(W3, W3B, DIN * HID);
  k_cast_bf16<<<(64 * DIN + 255) / 256, blk, 0, stream>>>(icW0, icW0B, 64 * DIN);
  k_cast_bf16<<<(64 * DIN + 255) / 256, blk, 0, stream>>>(icW1, icW1B, 64 * DIN);
  k_softplus_bf16<<<(64 * 64 + 255) / 256, blk, 0, stream>>>(icU0, spU0B, 64 * 64);

  // ---- Lm = V^T (eig V)  (1024x1024x1024) -> bf16 ----
  gemm_bf16<128, 128, M_STORE_BF16><<<dim3(DIMD / 128, DIMD / 128), blk, 0, stream>>>(
      PTB, DIMD, PEB, DIMD, DIMD, nullptr, nullptr, 0, nullptr, 0, nullptr, LmB, DIMD, 0);

  // ---- du += Hu @ Lm.T  (4096x1024x1024) into out cols [0,1024) ----
  gemm_bf16<128, 128, M_ADD_OUT><<<dim3(DIMD / 128, NB / 128), blk, 0, stream>>>(
      HuB, DIMD, LmB, DIMD, DIMD, nullptr, nullptr, 0, nullptr, 0, out, nullptr, LDO, 0);

  // ---- control net ----
  gemm_bf16<128, 128, M_TANH_BF16><<<dim3(HID / 128, NB / 128), blk, 0, stream>>>(
      stB, DIN, W1B, DIN, DIN, b1, nullptr, 0, nullptr, 0, nullptr, h1B, HID, 0);
  gemm_bf16<128, 128, M_TANH_BF16><<<dim3(HID / 128, NB / 128), blk, 0, stream>>>(
      h1B, HID, W2B, HID, HID, b2, nullptr, 0, nullptr, 0, nullptr, h2B, HID, 0);
  gemm_bf16<128, 128, M_CTRL><<<dim3(DIN / 128, NB / 128), blk, 0, stream>>>(
      h2B, HID, W3B, HID, HID, b3, nullptr, 0, stB, DIN, out, nullptr, LDO, 0);

  // ---- ICNN ----
  gemm_bf16<128, 64, M_SR_BF16><<<dim3(1, NB / 128), blk, 0, stream>>>(
      stateB, DIN, icW0B, DIN, DIN, icb0, nullptr, 0, nullptr, 0, nullptr, Z1B, 64, 0);
  gemm_bf16<128, 64, M_STORE_F32B><<<dim3(1, NB / 128), blk, 0, stream>>>(
      stateB, DIN, icW1B, DIN, DIN, icb1, nullptr, 0, nullptr, 0, T2f, nullptr, 64, 0);
  gemm_bf16<128, 64, M_ADDC_SR_F32><<<dim3(1, NB / 128), blk, 0, stream>>>(
      Z1B, 64, spU0B, 64, 64, nullptr, T2f, 64, nullptr, 0, Z2f, nullptr, 64, 0);

  // ---- tails ----
  k_z0<<<1, 64, 0, stream>>>(icb0, icb1, icb2, icU0, icU1, z0p);
  k_final<<<NB / 8, blk, 0, stream>>>(state, Z2f, icW2, icb2, icU1, z0p, out);
}
